// PerformerAttention_58463094833736
// MI455X (gfx1250) — matrix operand order repaired and checked by reading, not compiled, not run on hardware
//
#include <hip/hip_runtime.h>

// Problem sizes (fixed by the reference)
#define SEQ   2048
#define DM    1024
#define NH    16
#define HD    64
#define NF    256
#define BATCH 4
#define MTOT  (BATCH * SEQ)   // 8192 total rows

typedef __attribute__((ext_vector_type(16))) __bf16 v16bf;
typedef __attribute__((ext_vector_type(8)))  __bf16 v8bf;
typedef __attribute__((ext_vector_type(8)))  float  v8f;

static __device__ __forceinline__ int lane_id() { return threadIdx.x & 31; }

// ---- CDNA5 async global->LDS (ISA 08_async_tensor): per-lane 16B transfer,
// tracked with ASYNCcnt.  lds_off = low 32 bits of generic LDS pointer.
static __device__ __forceinline__ void async_ld_b128(unsigned lds_off,
                                                     const void* gptr) {
  unsigned long long ga = (unsigned long long)(uintptr_t)gptr;
  asm volatile("global_load_async_to_lds_b128 %0, %1, off"
               :: "v"(lds_off), "v"(ga) : "memory");
}
static __device__ __forceinline__ void wait_async_le10() {
  asm volatile("s_wait_asynccnt 10" ::: "memory");
}
static __device__ __forceinline__ void wait_async_le0() {
  asm volatile("s_wait_asynccnt 0" ::: "memory");
}

// A-fragment (16x32, bf16). Per ISA 7.12.2: lane m=l&15; lanes<16 hold K chunks
// {0..7, 16..23}; lanes>=16 hold {8..15, 24..31}.
static __device__ __forceinline__ v16bf load_fragA(const __bf16* p, int ld) {
  const int l  = lane_id();
  const int r  = l & 15;
  const int kb = (l >> 4) << 3;        // 0 or 8
  const __bf16* q = p + (size_t)r * ld + kb;
  v8bf lo = *(const v8bf*)(q);
  v8bf hi = *(const v8bf*)(q + 16);
  v16bf a;
#pragma unroll
  for (int i = 0; i < 8; ++i) { a[i] = lo[i]; a[i + 8] = hi[i]; }
  return a;
}

// B-fragment (32x16, bf16), source stored row-major as [n][k] (B[k,n]=src[n*ld+k]).
// Lane n=l&15; lanes<16 hold K chunks {0..7, 16..23}; lanes>=16 hold {8..15, 24..31}.
static __device__ __forceinline__ v16bf load_fragB(const __bf16* p, int ld) {
  const int l  = lane_id();
  const int n  = l & 15;
  const int kb = (l >> 4) << 3;        // 0 or 8
  const __bf16* q = p + (size_t)n * ld + kb;
  v8bf lo = *(const v8bf*)(q);
  v8bf hi = *(const v8bf*)(q + 16);
  v16bf b;
#pragma unroll
  for (int i = 0; i < 8; ++i) { b[i] = lo[i]; b[i + 8] = hi[i]; }
  return b;
}

static __device__ __forceinline__ v8f wmma_bf16(v16bf a, v16bf b, v8f c) {
  return __builtin_amdgcn_wmma_f32_16x16x32_bf16(false, a, false, b, (short)0, c,
                                                 false, false);
}

// One wave computes a 16(M) x 64(N) tile of C = A * W^T.
// K must be a multiple of 64 (true for all call sites: 1024 and 64).
// Ping-pong software pipeline: fragments for the next 32-wide K-step are issued
// before the WMMAs of the current step, with NO register copies (buffers swap
// roles each half-iteration), so waits cover a full step of load latency.
static __device__ __forceinline__ void gemm_tile_16x64(
    const __bf16* __restrict__ A, int lda, const __bf16* __restrict__ W, int ldw,
    int K, int row0, int n0, v8f acc[4]) {
  const __bf16* Ap = A + (size_t)row0 * lda;
  const __bf16* Wp = W + (size_t)n0 * ldw;

  v16bf a0 = load_fragA(Ap, lda), a1;
  v16bf b0[4], b1[4];
#pragma unroll
  for (int j = 0; j < 4; ++j)
    b0[j] = load_fragB(Wp + (size_t)(16 * j) * ldw, ldw);

  for (int k = 0; k + 64 <= K; k += 64) {
    // prefetch k+32, compute k
    a1 = load_fragA(Ap + k + 32, lda);
#pragma unroll
    for (int j = 0; j < 4; ++j)
      b1[j] = load_fragB(Wp + (size_t)(16 * j) * ldw + k + 32, ldw);
#pragma unroll
    for (int j = 0; j < 4; ++j) acc[j] = wmma_bf16(a0, b0[j], acc[j]);
    // prefetch k+64, compute k+32
    if (k + 64 < K) {
      a0 = load_fragA(Ap + k + 64, lda);
#pragma unroll
      for (int j = 0; j < 4; ++j)
        b0[j] = load_fragB(Wp + (size_t)(16 * j) * ldw + k + 64, ldw);
    }
#pragma unroll
    for (int j = 0; j < 4; ++j) acc[j] = wmma_bf16(a1, b1[j], acc[j]);
  }
}

__global__ void cvt_bf16_kernel(const float* __restrict__ src,
                                __bf16* __restrict__ dst, int n) {
  int i = blockIdx.x * blockDim.x + threadIdx.x;
  if (i < n) dst[i] = (__bf16)src[i];
}

// ---------------- QKV projection: q,k row-major bf16; v stored transposed [B,H,hd,S]
__global__ __launch_bounds__(256)
void qkv_kernel(const __bf16* __restrict__ xbf,
                const __bf16* __restrict__ wq, const __bf16* __restrict__ wk,
                const __bf16* __restrict__ wv,
                const float* __restrict__ bq, const float* __restrict__ bk,
                const float* __restrict__ bv,
                __bf16* __restrict__ qout, __bf16* __restrict__ kout,
                __bf16* __restrict__ vtout) {
  const int wave = threadIdx.x >> 5;
  const int l    = lane_id();
  const int row0 = blockIdx.x * 128 + wave * 16;
  const int n0   = blockIdx.y * 64;
  const int z    = blockIdx.z;

  const __bf16* W    = (z == 0) ? wq : (z == 1) ? wk : wv;
  const float*  bias = (z == 0) ? bq : (z == 1) ? bk : bv;

  v8f zero = {0.f, 0.f, 0.f, 0.f, 0.f, 0.f, 0.f, 0.f};
  v8f acc[4] = {zero, zero, zero, zero};
  gemm_tile_16x64(xbf, DM, W, DM, DM, row0, n0, acc);

  const int rowadd = (l >> 4) * 8;
  const int col_lo = l & 15;
#pragma unroll
  for (int j = 0; j < 4; ++j) {
    const int   col = n0 + 16 * j + col_lo;
    const float bv_ = bias[col];
#pragma unroll
    for (int r = 0; r < 8; ++r) {
      const int   m = row0 + r + rowadd;
      const float v = acc[j][r] + bv_;
      if (z == 0) {
        qout[(size_t)m * DM + col] = (__bf16)v;
      } else if (z == 1) {
        kout[(size_t)m * DM + col] = (__bf16)v;
      } else {
        const int b = m >> 11, s = m & (SEQ - 1);
        const int h = col >> 6, d = col & 63;
        vtout[((size_t)(b * NH + h) * HD + d) * SEQ + s] = (__bf16)v;
      }
    }
  }
}

// ---------------- Feature projection: qp = (q * P^T)/16, kp = k * P^T  (bf16, [BH,S,NF])
__global__ __launch_bounds__(256)
void feat_kernel(const __bf16* __restrict__ qbf, const __bf16* __restrict__ kbf,
                 const __bf16* __restrict__ pbf,
                 __bf16* __restrict__ qp, __bf16* __restrict__ kp) {
  const int wave = threadIdx.x >> 5;
  const int l    = lane_id();
  const int z    = blockIdx.z;
  const int sel  = z >> 6;           // 0 -> q, 1 -> k
  const int bh   = z & 63;
  const int b    = bh >> 4, h = bh & 15;
  const int row0 = blockIdx.x * 128 + wave * 16;   // within SEQ
  const int n0   = blockIdx.y * 64;                // within NF

  const __bf16* A = (sel ? kbf : qbf) + (size_t)b * SEQ * DM + h * HD;

  v8f zero = {0.f, 0.f, 0.f, 0.f, 0.f, 0.f, 0.f, 0.f};
  v8f acc[4] = {zero, zero, zero, zero};
  gemm_tile_16x64(A, DM, pbf, HD, HD, row0, n0, acc);

  const float scale = sel ? 1.0f : 0.0625f;  // fold 1/sqrt(NF) into qp
  __bf16* dst = (sel ? kp : qp) + (size_t)bh * SEQ * NF;
  const int rowadd = (l >> 4) * 8;
  const int col_lo = l & 15;
#pragma unroll
  for (int j = 0; j < 4; ++j)
#pragma unroll
    for (int r = 0; r < 8; ++r)
      dst[(size_t)(row0 + r + rowadd) * NF + n0 + 16 * j + col_lo] =
          (__bf16)(acc[j][r] * scale);
}

// ---------------- Flash attention over random features; output bf16 [B,S,D] layout
// Dynamic LDS layout (96 KB total):
//   [0,32K)   kp tile buf0   [32K,64K) kp tile buf1
//   [64K,72K) vt tile buf0   [72K,80K) vt tile buf1
//   [80K,96K) per-wave P staging (8 x 2KB)
__global__ __launch_bounds__(256)
void attn_kernel(const __bf16* __restrict__ qp, const __bf16* __restrict__ kp,
                 const __bf16* __restrict__ vt, __bf16* __restrict__ ao) {
  extern __shared__ __align__(16) char smem[];

  const int wave = threadIdx.x >> 5;
  const int l    = lane_id();
  const int tid  = threadIdx.x;
  const int bh   = blockIdx.y;
  const int b    = bh >> 4, h = bh & 15;
  const int row0 = blockIdx.x * 128 + wave * 16;   // within SEQ

  const __bf16* qp_bh = qp + (size_t)bh * SEQ * NF;
  const __bf16* kp_bh = kp + (size_t)bh * SEQ * NF;
  const __bf16* vt_bh = vt + (size_t)bh * HD * SEQ;
  __bf16* pw = (__bf16*)(smem + 81920) + wave * (16 * 64);

  // Issue async copies for one 64-wide t-block into buffer `buf` (0/1).
  // Per thread: 8 x b128 (kp, contiguous 32 KB) + 2 x b128 (vt, strided rows).
  auto prefetch = [&](int tblk, int buf) {
    const char* gk = (const char*)(kp_bh + (size_t)tblk * 64 * NF);
    char* lk = smem + buf * 32768;
#pragma unroll
    for (int u = 0; u < 8; ++u) {
      const int i = tid + u * 256;
      async_ld_b128((unsigned)(uintptr_t)(lk + i * 16), gk + i * 16);
    }
    const __bf16* gv = vt_bh + tblk * 64;
    char* lv = smem + 65536 + buf * 8192;
#pragma unroll
    for (int u = 0; u < 2; ++u) {
      const int i = tid + u * 256;
      const int dr = i >> 3, c8 = i & 7;
      async_ld_b128((unsigned)(uintptr_t)(lv + i * 16),
                    (const char*)(gv + (size_t)dr * SEQ + c8 * 8));
    }
  };

  // Keep this wave's 16x256 qp block resident as 8 A-fragments.
  v16bf aq[8];
#pragma unroll
  for (int kk = 0; kk < 8; ++kk)
    aq[kk] = load_fragA(qp_bh + (size_t)row0 * NF + kk * 32, NF);

  v8f zero = {0.f, 0.f, 0.f, 0.f, 0.f, 0.f, 0.f, 0.f};
  v8f out[4] = {zero, zero, zero, zero};
  float mrow[8], lrow[8];
#pragma unroll
  for (int r = 0; r < 8; ++r) { mrow[r] = -1e30f; lrow[r] = 0.f; }

  const int rowadd = (l >> 4) * 8;
  const int col_lo = l & 15;
  const int NT = SEQ / 64;   // 32 t-blocks

  prefetch(0, 0);
  for (int tb = 0; tb < NT; ++tb) {
    const int cur = tb & 1;
    if (tb + 1 < NT) {
      prefetch(tb + 1, cur ^ 1);
      wait_async_le10();   // 10 newest (next tile) may remain outstanding
    } else {
      wait_async_le0();
    }
    __syncthreads();       // all waves' copies for `cur` complete

    const __bf16* kpt = (const __bf16*)(smem + cur * 32768);
    const __bf16* vtt = (const __bf16*)(smem + 65536 + cur * 8192);

    // Scores: 16 x 64 block, K = 256 (1/sqrt(NF) already folded into qp).
    v8f sc[4] = {zero, zero, zero, zero};
#pragma unroll
    for (int kk = 0; kk < 8; ++kk) {
#pragma unroll
      for (int j = 0; j < 4; ++j) {
        v16bf bk = load_fragB(kpt + (size_t)(16 * j) * NF + kk * 32, NF);
        sc[j] = wmma_bf16(aq[kk], bk, sc[j]);
      }
    }

    // Online softmax: rows on element index r (+8 for upper-half lanes);
    // a row spans 16 lanes of a half -> xor-reduce with masks 1,2,4,8.
    float alpha[8];
#pragma unroll
    for (int r = 0; r < 8; ++r) {
      float v = fmaxf(fmaxf(sc[0][r], sc[1][r]), fmaxf(sc[2][r], sc[3][r]));
#pragma unroll
      for (int mk = 1; mk <= 8; mk <<= 1) v = fmaxf(v, __shfl_xor(v, mk, 32));
      const float mn = fmaxf(mrow[r], v);
      alpha[r] = __expf(mrow[r] - mn);
      mrow[r]  = mn;
    }
#pragma unroll
    for (int j = 0; j < 4; ++j)
#pragma unroll
      for (int r = 0; r < 8; ++r) sc[j][r] = __expf(sc[j][r] - mrow[r]);
#pragma unroll
    for (int r = 0; r < 8; ++r) {
      float v = sc[0][r] + sc[1][r] + sc[2][r] + sc[3][r];
#pragma unroll
      for (int mk = 1; mk <= 8; mk <<= 1) v += __shfl_xor(v, mk, 32);
      lrow[r] = alpha[r] * lrow[r] + v;
    }
#pragma unroll
    for (int j = 0; j < 4; ++j)
#pragma unroll
      for (int r = 0; r < 8; ++r) out[j][r] *= alpha[r];

    // Re-fragment P through the per-wave LDS tile (D-layout -> A-layout).
#pragma unroll
    for (int j = 0; j < 4; ++j)
#pragma unroll
      for (int r = 0; r < 8; ++r)
        pw[(size_t)(r + rowadd) * 64 + 16 * j + col_lo] = (__bf16)sc[j][r];

    // out(16x64 over d) += P(16x64 over t) * V(t x d), V from vT tile.
#pragma unroll
    for (int c = 0; c < 2; ++c) {
      v16bf ap = load_fragA(pw + c * 32, 64);
#pragma unroll
      for (int j2 = 0; j2 < 4; ++j2) {
        v16bf bv = load_fragB(vtt + (size_t)(16 * j2) * 64 + c * 32, 64);
        out[j2] = wmma_bf16(ap, bv, out[j2]);
      }
    }
    __syncthreads();       // all waves done reading `cur` before it is refilled
  }

  // Normalize and write to bf16 [B,S,D] for the output projection.
#pragma unroll
  for (int r = 0; r < 8; ++r) lrow[r] = 1.0f / lrow[r];
#pragma unroll
  for (int j2 = 0; j2 < 4; ++j2)
#pragma unroll
    for (int r = 0; r < 8; ++r) {
      const int m   = row0 + r + rowadd;
      const int col = h * HD + 16 * j2 + col_lo;
      ao[(size_t)(b * SEQ + m) * DM + col] = (__bf16)(out[j2][r] * lrow[r]);
    }
}

// ---------------- Output projection: fp32 result
__global__ __launch_bounds__(256)
void oproj_kernel(const __bf16* __restrict__ ao, const __bf16* __restrict__ wo,
                  const float* __restrict__ bo, float* __restrict__ out) {
  const int wave = threadIdx.x >> 5;
  const int l    = lane_id();
  const int row0 = blockIdx.x * 128 + wave * 16;
  const int n0   = blockIdx.y * 64;

  v8f zero = {0.f, 0.f, 0.f, 0.f, 0.f, 0.f, 0.f, 0.f};
  v8f acc[4] = {zero, zero, zero, zero};
  gemm_tile_16x64(ao, DM, wo, DM, DM, row0, n0, acc);

  const int rowadd = (l >> 4) * 8;
  const int col_lo = l & 15;
#pragma unroll
  for (int j = 0; j < 4; ++j) {
    const int   col = n0 + 16 * j + col_lo;
    const float bv_ = bo[col];
#pragma unroll
    for (int r = 0; r < 8; ++r)
      out[(size_t)(row0 + r + rowadd) * DM + col] = acc[j][r] + bv_;
  }
}

extern "C" void kernel_launch(void* const* d_in, const int* in_sizes, int n_in,
                              void* d_out, int out_size, void* d_ws, size_t ws_size,
                              hipStream_t stream) {
  (void)in_sizes; (void)n_in; (void)out_size; (void)ws_size;
  const float* x  = (const float*)d_in[0];
  const float* Wq = (const float*)d_in[1];
  const float* bq = (const float*)d_in[2];
  const float* Wk = (const float*)d_in[3];
  const float* bk = (const float*)d_in[4];
  const float* Wv = (const float*)d_in[5];
  const float* bv = (const float*)d_in[6];
  const float* Wo = (const float*)d_in[7];
  const float* bo = (const float*)d_in[8];
  const float* P  = (const float*)d_in[9];
  float* out = (float*)d_out;

  // Workspace carve-up (element offsets into a bf16 array; ~216 MiB total).
  __bf16* base  = (__bf16*)d_ws;
  __bf16* x_bf  = base;                         // 8388608
  __bf16* q_bf  = base + 8388608;               // 8388608
  __bf16* k_bf  = base + 16777216;              // 8388608
  __bf16* vt_bf = base + 25165824;              // 8388608  [B,H,hd,S]
  __bf16* ao_bf = base + 33554432;              // 8388608  [B,S,D]
  __bf16* wq_bf = base + 41943040;              // 1048576
  __bf16* wk_bf = base + 42991616;              // 1048576
  __bf16* wv_bf = base + 44040192;              // 1048576
  __bf16* wo_bf = base + 45088768;              // 1048576
  __bf16* p_bf  = base + 46137344;              // 16384
  __bf16* qp_bf = base + 46153728;              // 33554432 [BH,S,NF]
  __bf16* kp_bf = base + 79708160;              // 33554432

  auto cvt = [&](const float* s, __bf16* d, int n) {
    cvt_bf16_kernel<<<(n + 255) / 256, 256, 0, stream>>>(s, d, n);
  };
  cvt(x,  x_bf,  8388608);
  cvt(Wq, wq_bf, 1048576);
  cvt(Wk, wk_bf, 1048576);
  cvt(Wv, wv_bf, 1048576);
  cvt(Wo, wo_bf, 1048576);
  cvt(P,  p_bf,  16384);

  qkv_kernel<<<dim3(64, 16, 3), 256, 0, stream>>>(x_bf, wq_bf, wk_bf, wv_bf,
                                                  bq, bk, bv, q_bf, k_bf, vt_bf);
  feat_kernel<<<dim3(16, 4, 128), 256, 0, stream>>>(q_bf, k_bf, p_bf, qp_bf, kp_bf);
  attn_kernel<<<dim3(16, 64), 256, 98304, stream>>>(qp_bf, kp_bf, vt_bf, ao_bf);
  oproj_kernel<<<dim3(64, 16), 256, 0, stream>>>(ao_bf, wo_bf, bo, out);
}
